// VGG13_17626545783159
// MI455X (gfx1250) — compile-verified, matrix operand order checked by reading, not run on hardware
//
#include <hip/hip_runtime.h>
#include <hip/hip_bf16.h>
#include <math.h>

typedef __attribute__((ext_vector_type(8)))  _Float16 v8h;
typedef __attribute__((ext_vector_type(16))) _Float16 v16h;
typedef __attribute__((ext_vector_type(8)))  float    v8f;

#define EPSBN 1e-5f

// ---------------------------------------------------------------------------
// Input prep: feats (B,32,32,32,1) f32 + mask -> act0 (B,32,32,32,32ch) f16
// channel 0 = masked feat, channels 1..31 = 0. Also normalize mask to 0/1.
// ---------------------------------------------------------------------------
__global__ __launch_bounds__(256)
void prep_input_k(const float* __restrict__ feats,
                  const unsigned char* __restrict__ maskIn,
                  _Float16* __restrict__ act0,
                  unsigned char* __restrict__ m0, int total)
{
    int v = blockIdx.x * blockDim.x + threadIdx.x;
    if (v >= total) return;
    unsigned char m = maskIn[v] ? 1 : 0;
    m0[v] = m;
    v8h z = {};
    v8h first = z;
    first[0] = m ? (_Float16)feats[v] : (_Float16)0.f;
    v8h* p = (v8h*)(act0 + (size_t)v * 32);
    p[0] = first; p[1] = z; p[2] = z; p[3] = z;
}

// ---------------------------------------------------------------------------
// Weight prep: w (3,3,3,cin,cout) f32 -> f16 fragments in WMMA B layout.
// k = kpos*cinp + ci  (cin padded to cinp with zeros).
// Fragment element (lane, j) of block (nb, kb):
//   k = kb*32 + (j%8) + (j/8)*16 + (lane/16)*8 ; n = nb*16 + lane%16
// Stored flat at ((nb*nKblk + kb)*32 + lane)*16 + j  -> each lane's fragment
// is 32 contiguous bytes.
// ---------------------------------------------------------------------------
__global__ __launch_bounds__(256)
void prep_w_k(const float* __restrict__ w, _Float16* __restrict__ dst,
              int cin, int cinp, int cout, int nKblk, int total)
{
    int t = blockIdx.x * blockDim.x + threadIdx.x;
    if (t >= total) return;
    int j    = t & 15;
    int lane = (t >> 4) & 31;
    int frag = t >> 9;
    int kb = frag % nKblk;
    int nb = frag / nKblk;
    int k = (kb << 5) + (j & 7) + ((j >> 3) << 4) + ((lane >> 4) << 3);
    int n = (nb << 4) + (lane & 15);
    int kpos = k / cinp;
    int ci   = k - kpos * cinp;
    float val = (ci < cin) ? w[((size_t)(kpos * cin + ci)) * cout + n] : 0.f;
    dst[t] = (_Float16)val;
}

// ---------------------------------------------------------------------------
// Implicit-GEMM conv3d 3x3x3 SAME + fused BN/ReLU/mask (or split-K atomic
// accumulate). Templated on layer shape -> all addressing is shifts/ands,
// inner channel loop fully unrolled.
// One wave computes a (16*MT)(M voxels) x 64(N channels) tile: every B
// fragment is reused by MT WMMAs.
//   grid = (Mtotal/(16*MT) * COUT/64, S) ; blockIdx.y = tap chunk (split-K).
// A fragments: 2x b128 per tile straight from NDHWC f16 activations.
// B fragments: 4x 32B contiguous loads into distinct registers, then 4*MT
// back-to-back WMMAs (one wait per K-step instead of one per WMMA).
// ---------------------------------------------------------------------------
template<int G, int CINP, int COUT, int MT>
__global__ __launch_bounds__(32)
void conv_t(const _Float16* __restrict__ actIn,
            const _Float16* __restrict__ wgt,
            const unsigned char* __restrict__ maskB,
            const float* __restrict__ bnG, const float* __restrict__ bnB,
            const float* __restrict__ bnM, const float* __restrict__ bnV,
            _Float16* __restrict__ actOut, float* __restrict__ scratch,
            int kposChunk, int split)
{
    constexpr int NKB   = CINP / 32;        // K-blocks per tap
    constexpr int NKBLK = 27 * NKB;         // total K-blocks
    constexpr int NBLK  = COUT / 64;        // 64-channel N blocks
    constexpr int G2 = G * G, G3 = G2 * G;

    const int lane = threadIdx.x;
    const int nb64 = blockIdx.x % NBLK;     // NBLK is pow2 -> and
    const int mt   = blockIdx.x / NBLK;     // pow2 -> shift

    const int kp0 = blockIdx.y * kposChunk;
    int kp1 = kp0 + kposChunk;
    if (kp1 > 27) kp1 = 27;

    const int mBase = mt * (16 * MT);
    const int laneHalf = lane >> 4;

    int bc[MT], zc[MT], yc[MT], xc[MT];
    #pragma unroll
    for (int u = 0; u < MT; ++u) {
        int mLane = mBase + u * 16 + (lane & 15);
        bc[u] = mLane / G3;                  // pow2 -> shift
        int rem = mLane - bc[u] * G3;
        zc[u] = rem / G2; rem -= zc[u] * G2;
        yc[u] = rem / G;
        xc[u] = rem - yc[u] * G;
    }

    v8f acc[4 * MT] = {};
    constexpr size_t nbStride = (size_t)NKBLK * 512;    // halfs per 16-chan block
    const _Float16* wBase = wgt + (size_t)(nb64 * 4) * nbStride + (size_t)lane * 16;
    const _Float16* aBase = actIn + (size_t)laneHalf * 8;

    for (int kpos = kp0; kpos < kp1; ++kpos) {
        const int dz = kpos / 9, dyx = kpos - dz * 9;   // const divisors
        const int dy = dyx / 3, dx = dyx - dy * 3;

        const _Float16* ap[MT];
        bool inb[MT];
        #pragma unroll
        for (int u = 0; u < MT; ++u) {
            const int nz = zc[u] + dz - 1, ny = yc[u] + dy - 1, nx = xc[u] + dx - 1;
            inb[u] = ((unsigned)nz < (unsigned)G) &&
                     ((unsigned)ny < (unsigned)G) &&
                     ((unsigned)nx < (unsigned)G);
            ap[u] = aBase + (size_t)(((bc[u] * G + nz) * G + ny) * G + nx) * CINP;
        }
        const _Float16* wp = wBase + (size_t)(kpos * NKB) * 512;

        #pragma unroll
        for (int c = 0; c < NKB; ++c) {
            union AF { v16h v; v8h h[2]; };
            AF af[MT];
            v8h zero8 = {};
            #pragma unroll
            for (int u = 0; u < MT; ++u) {
                af[u].h[0] = zero8; af[u].h[1] = zero8;
                if (inb[u]) {
                    const v8h* a8 = (const v8h*)(ap[u] + c * 32);
                    af[u].h[0] = a8[0];   // K {0..7}  (lane<16) / {8..15}
                    af[u].h[1] = a8[2];   // K {16..23}          / {24..31}
                }
            }
            v16h bf[4];
            #pragma unroll
            for (int t = 0; t < 4; ++t)
                bf[t] = *(const v16h*)(wp + (size_t)c * 512 + (size_t)t * nbStride);
            #pragma unroll
            for (int t = 0; t < 4; ++t)
                #pragma unroll
                for (int u = 0; u < MT; ++u)
                    acc[u * 4 + t] = __builtin_amdgcn_wmma_f32_16x16x32_f16(
                        false, af[u].v, false, bf[t], (short)0, acc[u * 4 + t],
                        false, false);
        }
    }

    // D layout: VGPR r -> M = r + 8*(lane>=16), N = lane%16
    const int nTileBase = nb64 * 64;
    if (split) {
        #pragma unroll
        for (int t = 0; t < 4; ++t) {
            const int n = nTileBase + t * 16 + (lane & 15);
            #pragma unroll
            for (int u = 0; u < MT; ++u)
                #pragma unroll
                for (int rr = 0; rr < 8; ++rr) {
                    const int vox = mBase + u * 16 + (laneHalf << 3) + rr;
                    atomicAdd(scratch + (size_t)vox * COUT + n, acc[u * 4 + t][rr]);
                }
        }
    } else {
        #pragma unroll
        for (int t = 0; t < 4; ++t) {
            const int n = nTileBase + t * 16 + (lane & 15);
            const float gg = bnG[n] * rsqrtf(bnV[n] + EPSBN);
            const float bb = bnB[n] - bnM[n] * gg;
            #pragma unroll
            for (int u = 0; u < MT; ++u)
                #pragma unroll
                for (int rr = 0; rr < 8; ++rr) {
                    const int vox = mBase + u * 16 + (laneHalf << 3) + rr;
                    float val = acc[u * 4 + t][rr] * gg + bb;
                    val = maskB[vox] ? fmaxf(val, 0.f) : 0.f;
                    actOut[(size_t)vox * COUT + n] = (_Float16)val;
                }
        }
    }
}

// BN/ReLU/mask epilogue for split-K layers (reads f32 scratch)
__global__ __launch_bounds__(256)
void bn_epi_k(const float* __restrict__ scratch,
              const unsigned char* __restrict__ maskB,
              const float* __restrict__ bnG, const float* __restrict__ bnB,
              const float* __restrict__ bnM, const float* __restrict__ bnV,
              _Float16* __restrict__ actOut, int cout, int total)
{
    int idx = blockIdx.x * blockDim.x + threadIdx.x;
    if (idx >= total) return;
    int n = idx & (cout - 1);        // cout is pow2
    int vox = idx / cout;
    float gg = bnG[n] * rsqrtf(bnV[n] + EPSBN);
    float val = scratch[idx] * gg + (bnB[n] - bnM[n] * gg);
    actOut[idx] = (_Float16)(maskB[vox] ? fmaxf(val, 0.f) : 0.f);
}

// 2x2x2 masked max-pool. Post-ReLU masked activations are >=0 and zero at
// unoccupied voxels, so pooled value == plain max over the 8 children.
__global__ __launch_bounds__(256)
void pool_k(const _Float16* __restrict__ in, _Float16* __restrict__ out,
            const unsigned char* __restrict__ mIn, unsigned char* __restrict__ mOut,
            int Gi, int C, int total)
{
    int idx = blockIdx.x * blockDim.x + threadIdx.x;
    if (idx >= total) return;
    int c = idx & (C - 1);           // C is pow2
    int vo = idx / C;
    int Go = Gi >> 1;
    int Go3 = Go * Go * Go;
    int b = vo / Go3; int r = vo - b * Go3;
    int zo = r / (Go * Go); r -= zo * Go * Go;
    int yo = r / Go;
    int xo = r - yo * Go;
    float mx = 0.f;
    int occ = 0;
    #pragma unroll
    for (int dz = 0; dz < 2; ++dz)
    #pragma unroll
    for (int dy = 0; dy < 2; ++dy)
    #pragma unroll
    for (int dx = 0; dx < 2; ++dx) {
        int vi = (((b * Gi + 2 * zo + dz) * Gi + 2 * yo + dy) * Gi + 2 * xo + dx);
        occ |= mIn[vi];
        mx = fmaxf(mx, (float)in[(size_t)vi * C + c]);
    }
    out[(size_t)vo * C + c] = (_Float16)mx;
    if (c == 0) mOut[vo] = occ ? 1 : 0;
}

// Masked global mean over 2^3 voxels + L2 normalize. One block per batch.
__global__ __launch_bounds__(512)
void gpool_norm_k(const _Float16* __restrict__ act,
                  const unsigned char* __restrict__ m2,
                  float* __restrict__ out)
{
    __shared__ float red[512];
    int b = blockIdx.x;
    int c = threadIdx.x;
    float s = 0.f;
    #pragma unroll
    for (int i = 0; i < 8; ++i) s += (float)act[(size_t)(b * 8 + i) * 512 + c];
    int cnt = 0;
    #pragma unroll
    for (int i = 0; i < 8; ++i) cnt += m2[b * 8 + i];
    float emb = s / fmaxf((float)cnt, 1.f);
    red[c] = emb * emb;
    __syncthreads();
    for (int st = 256; st > 0; st >>= 1) {
        if (c < st) red[c] += red[c + st];
        __syncthreads();
    }
    float nrm = sqrtf(red[0]);
    out[b * 512 + c] = emb / fmaxf(nrm, 1e-12f);
}

// ---------------------------------------------------------------------------
struct ConvArgs {
    const _Float16* actIn; const _Float16* wgt; const unsigned char* mask;
    const float *bg, *bb, *bm, *bv;
    _Float16* actOut; float* scratch;
};

template<int G, int CINP, int COUT, int MT>
static inline void launch_conv(const ConvArgs& a, int S, int kposChunk, int split,
                               hipStream_t stream)
{
    const int Mtiles = 2 * G * G * G / (16 * MT);
    dim3 grid(Mtiles * (COUT / 64), S);
    conv_t<G, CINP, COUT, MT><<<grid, 32, 0, stream>>>(
        a.actIn, a.wgt, a.mask, a.bg, a.bb, a.bm, a.bv, a.actOut, a.scratch,
        kposChunk, split);
}

extern "C" void kernel_launch(void* const* d_in, const int* in_sizes, int n_in,
                              void* d_out, int out_size, void* d_ws, size_t ws_size,
                              hipStream_t stream)
{
    (void)in_sizes; (void)n_in; (void)out_size; (void)ws_size;
    const float* feats = (const float*)d_in[0];
    const unsigned char* maskIn = (const unsigned char*)d_in[1];  // numpy bool = 1 byte

    static const int GS[10]   = {32,32,16,16,8,8,4,4,2,2};
    static const int CINR[10] = {1,64,64,128,128,256,256,512,512,512};
    static const int CINP[10] = {32,64,64,128,128,256,256,512,512,512};
    static const int COUT[10] = {64,64,128,128,256,256,512,512,512,512};
    static const int SPL[10]  = {1,1,1,1,4,4,9,9,27,27};   // split over 27 taps

    // ---- carve workspace (~74 MB) ----
    char* base = (char*)d_ws;
    size_t off = 0;
    auto carve = [&](size_t bytes) -> char* {
        char* p = base + off;
        off = (off + bytes + 255) & ~(size_t)255;
        return p;
    };
    _Float16* wgt[10];
    for (int i = 0; i < 10; ++i)
        wgt[i] = (_Float16*)carve((size_t)27 * CINP[i] * COUT[i] * sizeof(_Float16));
    _Float16* actA = (_Float16*)carve((size_t)65536 * 64 * sizeof(_Float16));
    _Float16* actB = (_Float16*)carve((size_t)65536 * 64 * sizeof(_Float16));
    static const int MSZ[5] = {65536, 8192, 1024, 128, 16};
    unsigned char* maskL[5];
    for (int l = 0; l < 5; ++l) maskL[l] = (unsigned char*)carve(MSZ[l]);
    float* scratch = (float*)carve((size_t)128 * 512 * sizeof(float));

    // ---- prep ----
    prep_input_k<<<256, 256, 0, stream>>>(feats, maskIn, actA, maskL[0], 65536);
    for (int i = 0; i < 10; ++i) {
        int nKblk = 27 * CINP[i] / 32;
        int total = (COUT[i] / 16) * nKblk * 512;
        prep_w_k<<<(total + 255) / 256, 256, 0, stream>>>(
            (const float*)d_in[2 + 5 * i], wgt[i], CINR[i], CINP[i], COUT[i], nKblk, total);
    }

    // ---- layers ----
    _Float16* cur = actA;
    _Float16* nxt = actB;
    int level = 0;
    for (int i = 0; i < 10; ++i) {
        const int G = GS[i], cout = COUT[i];
        const int Mtotal = 2 * G * G * G;
        const int S = SPL[i];
        const int chunk = (27 + S - 1) / S;
        const int split = (S > 1) ? 1 : 0;
        ConvArgs a;
        a.actIn = cur; a.wgt = wgt[i]; a.mask = maskL[level];
        a.bg = (const float*)d_in[3 + 5 * i];
        a.bb = (const float*)d_in[4 + 5 * i];
        a.bm = (const float*)d_in[5 + 5 * i];
        a.bv = (const float*)d_in[6 + 5 * i];
        a.actOut = nxt; a.scratch = scratch;

        if (split)
            hipMemsetAsync(scratch, 0, (size_t)Mtotal * cout * sizeof(float), stream);

        switch (i) {
            case 0: launch_conv<32,  32,  64, 2>(a, S, chunk, split, stream); break;
            case 1: launch_conv<32,  64,  64, 2>(a, S, chunk, split, stream); break;
            case 2: launch_conv<16,  64, 128, 2>(a, S, chunk, split, stream); break;
            case 3: launch_conv<16, 128, 128, 2>(a, S, chunk, split, stream); break;
            case 4: launch_conv< 8, 128, 256, 2>(a, S, chunk, split, stream); break;
            case 5: launch_conv< 8, 256, 256, 2>(a, S, chunk, split, stream); break;
            case 6: launch_conv< 4, 256, 512, 2>(a, S, chunk, split, stream); break;
            case 7: launch_conv< 4, 512, 512, 2>(a, S, chunk, split, stream); break;
            case 8: launch_conv< 2, 512, 512, 1>(a, S, chunk, split, stream); break;
            case 9: launch_conv< 2, 512, 512, 1>(a, S, chunk, split, stream); break;
        }

        if (split) {
            int tot = Mtotal * cout;
            bn_epi_k<<<(tot + 255) / 256, 256, 0, stream>>>(
                scratch, maskL[level], a.bg, a.bb, a.bm, a.bv, nxt, cout, tot);
        }
        { _Float16* t = cur; cur = nxt; nxt = t; }

        if (i == 1 || i == 3 || i == 5 || i == 7) {
            int Go = G / 2;
            int tot = 2 * Go * Go * Go * cout;
            pool_k<<<(tot + 255) / 256, 256, 0, stream>>>(
                cur, nxt, maskL[level], maskL[level + 1], G, cout, tot);
            { _Float16* t = cur; cur = nxt; nxt = t; }
            level++;
        }
    }

    gpool_norm_k<<<2, 512, 0, stream>>>(cur, maskL[4], (float*)d_out);
}
